// FlashAttention3_64828236366195
// MI455X (gfx1250) — matrix operand order repaired and checked by reading, not compiled, not run on hardware
//
#include <hip/hip_runtime.h>
#include <hip/hip_bf16.h>

typedef __attribute__((ext_vector_type(16))) _Float16 v16h;
typedef __attribute__((ext_vector_type(8)))  _Float16 v8h;
typedef __attribute__((ext_vector_type(8)))  float    v8f;

#define B_    2
#define S_    2048
#define H_    16
#define D_    64
#define QTILE 128    // query rows per workgroup (8 waves x 16 rows)
#define KTILE 32     // keys per inner block (matches WMMA K=32 for PV)
#define NEGBIG (-1e9f)

union U16 { v16h v; v8h h[2]; };

static __device__ __forceinline__ v8f wmma_f16(v16h a, v16h b, v8f c) {
    return __builtin_amdgcn_wmma_f32_16x16x32_f16(false, a, false, b, (short)0, c, false, false);
}

__global__ __launch_bounds__(256)
void FlashAttention3_64828236366195_kernel(const float* __restrict__ q,
                                           const float* __restrict__ k,
                                           const float* __restrict__ v,
                                           const int*   __restrict__ mask,
                                           float* __restrict__ out)
{
    __shared__ _Float16 Ks[2][KTILE * D_];   // [key][kdim], f16
    __shared__ _Float16 Vt[2][D_ * KTILE];   // [d][key],   f16 (transposed)
    __shared__ float    Mb[2][KTILE];        // additive bias: 0 or -1e9
    __shared__ _Float16 Ps[8][16 * KTILE];   // per-wave P scratch [m][kkey]

    const int tid  = threadIdx.x;
    const int w    = tid >> 5;        // wave 0..7
    const int l    = tid & 31;        // lane in wave32
    const int n    = l & 15;          // column index / row-for-A
    const int half = l >> 4;          // 0: lanes 0-15, 1: lanes 16-31

    const int qt = blockIdx.x;        // 0..15
    const int h  = blockIdx.y;
    const int b  = blockIdx.z;
    const int q0 = qt * QTILE;
    const int nkb   = (qt + 1) * (QTILE / KTILE);          // causal key blocks
    const int nFull = (qt == 0) ? nkb : qt * (QTILE / KTILE); // fully-allowed blocks

    const float scale = 0.125f;       // 1/sqrt(64)

    // ---------------- Q A-fragments (normalize, scale, f16) ----------------
    const int qi_row = q0 + w * 16 + n;       // this lane's A-matrix row (M = n)
    const float* qrow = q + (((size_t)b * S_ + qi_row) * H_ + h) * D_;
    float ss = 0.f;
    #pragma unroll
    for (int kd = 0; kd < D_; ++kd) { float x = qrow[kd]; ss += x * x; }
    const float qmul = scale / fmaxf(sqrtf(ss), 1e-12f);
    v16h a_lo, a_hi;                           // K = 0..31 and 32..63
    #pragma unroll
    for (int t = 0; t < 16; ++t) {
        int kk = t + half * 16;   // contiguous K layout
        a_lo[t] = (_Float16)(qrow[kk]      * qmul);
        a_hi[t] = (_Float16)(qrow[kk + 32] * qmul);
    }

    // ---------------- running softmax state ----------------
    float m_run[8], l_run[8];
    v8f acc[4];
    #pragma unroll
    for (int i = 0; i < 8; ++i) { m_run[i] = NEGBIG; l_run[i] = 0.f; }
    #pragma unroll
    for (int dt = 0; dt < 4; ++dt)
        #pragma unroll
        for (int i = 0; i < 8; ++i) acc[dt][i] = 0.f;

    // ---------------- stage helper (vectorized) ----------------
    const int skk  = tid >> 3;            // key row 0..31
    const int skd0 = (tid & 7) << 3;      // dim 0,8,...,56
    auto stage = [&](int kb, int buf) {
        const int keyBase = kb * KTILE;
        size_t g = (((size_t)b * S_ + keyBase + skk) * H_ + h) * D_ + skd0;
        const float4* kp = (const float4*)(k + g);
        const float4* vp = (const float4*)(v + g);
        float4 k0 = kp[0], k1 = kp[1];
        float4 v0 = vp[0], v1 = vp[1];
        v8h kh;
        kh[0] = (_Float16)k0.x; kh[1] = (_Float16)k0.y;
        kh[2] = (_Float16)k0.z; kh[3] = (_Float16)k0.w;
        kh[4] = (_Float16)k1.x; kh[5] = (_Float16)k1.y;
        kh[6] = (_Float16)k1.z; kh[7] = (_Float16)k1.w;
        *(v8h*)&Ks[buf][skk * D_ + skd0] = kh;     // one ds_store_b128
        _Float16 vh[8];
        vh[0] = (_Float16)v0.x; vh[1] = (_Float16)v0.y;
        vh[2] = (_Float16)v0.z; vh[3] = (_Float16)v0.w;
        vh[4] = (_Float16)v1.x; vh[5] = (_Float16)v1.y;
        vh[6] = (_Float16)v1.z; vh[7] = (_Float16)v1.w;
        #pragma unroll
        for (int j = 0; j < 8; ++j)
            Vt[buf][(skd0 + j) * KTILE + skk] = vh[j];   // transpose scatter
        if (tid < KTILE)
            Mb[buf][tid] = mask[(size_t)b * S_ + keyBase + tid] ? 0.f : NEGBIG;
    };

    stage(0, 0);
    __syncthreads();

    for (int kb = 0; kb < nkb; ++kb) {
        const int buf = kb & 1;
        const int keyBase = kb * KTILE;
        const bool causal = (kb >= nFull);     // uniform across workgroup
        if (kb + 1 < nkb) stage(kb + 1, buf ^ 1);

        // ---- scores: S = Qn x K^T + key-mask bias (bias folded into C) ----
        const float bias0 = Mb[buf][n];
        const float bias1 = Mb[buf][16 + n];
        v8f c0, c1;
        #pragma unroll
        for (int nb = 0; nb < 2; ++nb) {
            U16 blo, bhi;
            const _Float16* kr = &Ks[buf][(nb * 16 + n) * D_];
            blo.h[0] = *(const v8h*)(kr + half * 16);
            blo.h[1] = *(const v8h*)(kr + half * 16 + 8);
            bhi.h[0] = *(const v8h*)(kr + 32 + half * 16);
            bhi.h[1] = *(const v8h*)(kr + 32 + half * 16 + 8);
            const float bb = (nb == 0) ? bias0 : bias1;
            v8f z;
            #pragma unroll
            for (int i = 0; i < 8; ++i) z[i] = bb;   // mask bias as C init
            z = wmma_f16(a_lo, blo.v, z);
            z = wmma_f16(a_hi, bhi.v, z);
            if (nb == 0) c0 = z; else c1 = z;
        }

        // ---- causal select only on diagonal blocks ----
        float s0[8], s1[8];
        if (causal) {
            const int ki0 = keyBase + n;
            const int ki1 = keyBase + 16 + n;
            #pragma unroll
            for (int i = 0; i < 8; ++i) {
                int qi = q0 + w * 16 + half * 8 + i;
                s0[i] = (ki0 <= qi) ? c0[i] : NEGBIG;
                s1[i] = (ki1 <= qi) ? c1[i] : NEGBIG;
            }
        } else {
            #pragma unroll
            for (int i = 0; i < 8; ++i) { s0[i] = c0[i]; s1[i] = c1[i]; }
        }

        // ---- online softmax (rows striped across 16 lanes) ----
        float bm[8];
        #pragma unroll
        for (int i = 0; i < 8; ++i) bm[i] = fmaxf(s0[i], s1[i]);
        #pragma unroll
        for (int off = 1; off < 16; off <<= 1)
            #pragma unroll
            for (int i = 0; i < 8; ++i)
                bm[i] = fmaxf(bm[i], __shfl_xor(bm[i], off, 32));

        float al[8], p0[8], p1[8], rs[8];
        #pragma unroll
        for (int i = 0; i < 8; ++i) {
            float mn = fmaxf(m_run[i], bm[i]);
            al[i] = __expf(m_run[i] - mn);
            p0[i] = __expf(s0[i] - mn);
            p1[i] = __expf(s1[i] - mn);
            rs[i] = p0[i] + p1[i];
            m_run[i] = mn;
        }
        #pragma unroll
        for (int off = 1; off < 16; off <<= 1)
            #pragma unroll
            for (int i = 0; i < 8; ++i)
                rs[i] += __shfl_xor(rs[i], off, 32);
        #pragma unroll
        for (int i = 0; i < 8; ++i) l_run[i] = l_run[i] * al[i] + rs[i];
        #pragma unroll
        for (int dt = 0; dt < 4; ++dt)
            #pragma unroll
            for (int i = 0; i < 8; ++i) acc[dt][i] *= al[i];

        // ---- transpose P (C layout -> A layout) through per-wave LDS ----
        _Float16* pw = Ps[w];
        #pragma unroll
        for (int i = 0; i < 8; ++i) {
            int m = i + half * 8;
            pw[m * KTILE + n]      = (_Float16)p0[i];
            pw[m * KTILE + 16 + n] = (_Float16)p1[i];
        }
        asm volatile("s_wait_dscnt 0" ::: "memory");  // same-wave cross-lane LDS RAW
        U16 pa;
        const _Float16* pr = &pw[n * KTILE];
        pa.h[0] = *(const v8h*)(pr + half * 16);       // K = half*16 + 0..7
        pa.h[1] = *(const v8h*)(pr + half * 16 + 8);   // K = half*16 + 8..15

        // ---- PV: acc += P (16x32) x V (32x64), 4 d-tiles ----
        #pragma unroll
        for (int dt = 0; dt < 4; ++dt) {
            U16 vb;
            const _Float16* vr = &Vt[buf][(dt * 16 + n) * KTILE + half * 16];
            vb.h[0] = *(const v8h*)(vr);
            vb.h[1] = *(const v8h*)(vr + 8);
            acc[dt] = wmma_f16(pa.v, vb.v, acc[dt]);
        }

        __syncthreads();   // next buffer staged; this buffer free
    }

    // ---------------- epilogue: out = acc / l ----------------
    #pragma unroll
    for (int i = 0; i < 8; ++i) {
        float invl = 1.0f / l_run[i];
        int qi = q0 + w * 16 + half * 8 + i;
        float* orow = out + (((size_t)b * S_ + qi) * H_ + h) * D_;
        #pragma unroll
        for (int dt = 0; dt < 4; ++dt)
            orow[dt * 16 + n] = acc[dt][i] * invl;
    }
}

extern "C" void kernel_launch(void* const* d_in, const int* in_sizes, int n_in,
                              void* d_out, int out_size, void* d_ws, size_t ws_size,
                              hipStream_t stream) {
    (void)in_sizes; (void)n_in; (void)out_size; (void)d_ws; (void)ws_size;
    const float* q   = (const float*)d_in[0];
    const float* k   = (const float*)d_in[1];
    const float* v   = (const float*)d_in[2];
    const int*   msk = (const int*)d_in[3];
    float* out = (float*)d_out;
    dim3 grid(S_ / QTILE, H_, B_);
    FlashAttention3_64828236366195_kernel<<<grid, dim3(256), 0, stream>>>(q, k, v, msk, out);
}
